// TA_42219528519914
// MI455X (gfx1250) — compile-verified, matrix operand order checked by reading, not run on hardware
//
#include <hip/hip_runtime.h>

#define NBATCH 4
#define NVAR   16
#define NLSEQ  1024
#define NCH    256
#define NAH    256
#define BMP    64     // rows per block, projection pass
#define BMA    128    // rows per block, attention pass

typedef __attribute__((ext_vector_type(16))) _Float16 v16h;
typedef __attribute__((ext_vector_type(8)))  _Float16 v8h;
typedef __attribute__((ext_vector_type(8)))  float    v8f;
typedef __attribute__((ext_vector_type(4)))  float    v4f;

// Load one 16x32 f16 WMMA operand fragment. Per CDNA5 layout, lane (0-15|16-31)
// needs k = [k0+h*8, +8) and [k0+16+h*8, +8) of its row: two 16B loads.
__device__ __forceinline__ v16h ldfrag(const _Float16* rowp, int k0, int hlf) {
  union { v16h v; v8h h[2]; } u;
  u.h[0] = *(const v8h*)(rowp + k0 + hlf * 8);
  u.h[1] = *(const v8h*)(rowp + k0 + 16 + hlf * 8);
  return u.v;
}

__device__ __forceinline__ v8f wmma_f16(v16h a, v16h b, v8f c) {
  return __builtin_amdgcn_wmma_f32_16x16x32_f16(false, a, false, b, (short)0, c,
                                                false, false);
}

// ---------------- pass 0: fc_w fp32 -> f16 (row-major [C][CA]) ----------------
__global__ __launch_bounds__(256) void cvt_fcw_kernel(const float* __restrict__ w,
                                                      _Float16* __restrict__ wh) {
  int i = blockIdx.x * 256 + threadIdx.x;
  wh[i] = (_Float16)w[i];
}

// ---------------- pass 1: Q/K/V projections ----------------
// Block: 256 thr = 8 waves. 4 row-strips of 16; wave w -> strip w>>1, n-half w&1.
__global__ __launch_bounds__(256) void proj_kernel(
    const float* __restrict__ X,
    const float* __restrict__ Qw, const float* __restrict__ Kw,
    const float* __restrict__ Vw,
    const float* __restrict__ Qb, const float* __restrict__ Kb,
    const float* __restrict__ Vb,
    _Float16* __restrict__ Qh, _Float16* __restrict__ Kh,
    _Float16* __restrict__ Vt)
{
  __shared__ _Float16 Xl[BMP][NCH];   // 32 KB  X tile, f16
  __shared__ _Float16 Wl[NCH][32];    // 16 KB  weight chunk, transposed [a][k]

  const int bv   = blockIdx.y;                 // b*V + v
  const int rb   = blockIdx.x;                 // row block (0..15)
  const int v    = bv & (NVAR - 1);
  const int tid  = threadIdx.x;
  const int lane = tid & 31, wv = tid >> 5;
  const int hlf  = lane >> 4, lc = lane & 15;
  const int strip = wv >> 1, nh = wv & 1;      // 8 n-tiles per wave

  const float* Xblk = X + ((size_t)bv * NLSEQ + (size_t)rb * BMP) * NCH;

  // stage X tile: issue all 16 b128 loads first (one loadcnt wait), then cvt+store
  {
    v4f xr[16];
    #pragma unroll
    for (int j = 0; j < 16; ++j) {
      int i = j * 256 + tid;
      int r = i >> 6, c = (i & 63) << 2;
      xr[j] = *(const v4f*)(Xblk + (size_t)r * NCH + c);
    }
    #pragma unroll
    for (int j = 0; j < 16; ++j) {
      int i = j * 256 + tid;
      int r = i >> 6, c = (i & 63) << 2;
      Xl[r][c + 0] = (_Float16)xr[j][0];
      Xl[r][c + 1] = (_Float16)xr[j][1];
      Xl[r][c + 2] = (_Float16)xr[j][2];
      Xl[r][c + 3] = (_Float16)xr[j][3];
    }
  }
  __syncthreads();

  const float* Wm[3] = { Qw + (size_t)v * NCH * NAH,
                         Kw + (size_t)v * NCH * NAH,
                         Vw + (size_t)v * NCH * NAH };
  const float* bm[3] = { Qb, Kb, Vb };

  #pragma unroll
  for (int m = 0; m < 3; ++m) {
    v8f acc[8];
    #pragma unroll
    for (int n = 0; n < 8; ++n) {
      float bb = bm[m][(nh * 8 + n) * 16 + lc];   // bias folded into C accum
      #pragma unroll
      for (int r = 0; r < 8; ++r) acc[n][r] = bb;
    }
    const float* W = Wm[m];
    for (int k0 = 0; k0 < NCH; k0 += 32) {
      __syncthreads();
      {  // transpose-stage weight chunk: 8 batched loads, then cvt+store
        v4f wr[8];
        #pragma unroll
        for (int j = 0; j < 8; ++j) {
          int i = j * 256 + tid;
          int k = i >> 6, c = (i & 63) << 2;
          wr[j] = *(const v4f*)(W + (size_t)(k0 + k) * NAH + c);
        }
        #pragma unroll
        for (int j = 0; j < 8; ++j) {
          int i = j * 256 + tid;
          int k = i >> 6, c = (i & 63) << 2;
          Wl[c + 0][k] = (_Float16)wr[j][0];
          Wl[c + 1][k] = (_Float16)wr[j][1];
          Wl[c + 2][k] = (_Float16)wr[j][2];
          Wl[c + 3][k] = (_Float16)wr[j][3];
        }
      }
      __syncthreads();
      v16h a = ldfrag(&Xl[strip * 16 + lc][0], k0, hlf);
      #pragma unroll
      for (int n = 0; n < 8; ++n) {
        v16h b = ldfrag(&Wl[(nh * 8 + n) * 16 + lc][0], 0, hlf);
        acc[n] = wmma_f16(a, b, acc[n]);
      }
    }
    const int grow = rb * BMP + strip * 16 + hlf * 8;
    if (m == 2) {                       // V stored transposed: Vt[bv][a][l]
      _Float16* dst = Vt + (size_t)bv * NAH * NLSEQ;
      #pragma unroll
      for (int n = 0; n < 8; ++n) {
        int a_ = (nh * 8 + n) * 16 + lc;
        union { v8h v; _Float16 e[8]; } pk;      // 8 consecutive l -> one b128
        #pragma unroll
        for (int r = 0; r < 8; ++r) pk.e[r] = (_Float16)acc[n][r];
        *(v8h*)(dst + (size_t)a_ * NLSEQ + grow) = pk.v;
      }
    } else {                            // Q/K row-major: [bv][l][a]
      _Float16* dst = (m == 0 ? Qh : Kh) + (size_t)bv * NLSEQ * NAH;
      #pragma unroll
      for (int n = 0; n < 8; ++n)
        #pragma unroll
        for (int r = 0; r < 8; ++r)
          dst[(size_t)(grow + r) * NAH + (nh * 8 + n) * 16 + lc] =
              (_Float16)acc[n][r];
    }
  }
}

// ---------------- pass 2: causal flash attention + fc + residual ----------------
__global__ __launch_bounds__(256) void attn_kernel(
    const float* __restrict__ X,
    const _Float16* __restrict__ Qh, const _Float16* __restrict__ Kh,
    const _Float16* __restrict__ Vt, const _Float16* __restrict__ fch,
    const float* __restrict__ fcb, const float* __restrict__ gama,
    float* __restrict__ out)
{
  __shared__ _Float16 stage[8][16 * 128];   // 32 KB: per-wave staging (P and Va)

  const int bv = blockIdx.y, rb = blockIdx.x;
  const int tid = threadIdx.x, lane = tid & 31, wv = tid >> 5;
  const int hlf = lane >> 4, lc = lane & 15;
  const int rowBase = rb * BMA + wv * 16;        // global query row of this wave

  const size_t bvo = (size_t)bv * NLSEQ * NAH;
  const _Float16* Kb_ = Kh + bvo;
  const _Float16* Vtb = Vt + (size_t)bv * NAH * NLSEQ;

  v16h qf[8];                                     // whole Q strip as A-frags
  {
    const _Float16* Qrow = Qh + bvo + (size_t)(rowBase + lc) * NAH;
    #pragma unroll
    for (int k = 0; k < 8; ++k) qf[k] = ldfrag(Qrow, k * 32, hlf);
  }

  v8f o[16];
  #pragma unroll
  for (int n = 0; n < 16; ++n)
    #pragma unroll
    for (int r = 0; r < 8; ++r) o[n][r] = 0.0f;

  float mrow[8], lrow[8];
  #pragma unroll
  for (int r = 0; r < 8; ++r) { mrow[r] = -1e30f; lrow[r] = 0.0f; }

  _Float16* Pst = &stage[wv][0];                  // 16x32 P bounce buffer

  const int ncb = ((rowBase + 15) >> 5) + 1;      // causal 32-key blocks
  for (int cb = 0; cb < ncb; ++cb) {
    const int col0 = cb * 32;
    const _Float16* K0 = Kb_ + (size_t)(col0 + lc) * NAH;
    const _Float16* K1 = K0 + (size_t)16 * NAH;
    if (cb + 1 < ncb) {                           // prefetch next key block
      __builtin_prefetch(K0 + (size_t)32 * NAH, 0, 1);
      __builtin_prefetch(K1 + (size_t)32 * NAH, 0, 1);
    }
    v8f s0, s1;
    #pragma unroll
    for (int r = 0; r < 8; ++r) { s0[r] = 0.0f; s1[r] = 0.0f; }
    #pragma unroll
    for (int k = 0; k < 8; ++k) {
      s0 = wmma_f16(qf[k], ldfrag(K0, k * 32, hlf), s0);
      s1 = wmma_f16(qf[k], ldfrag(K1, k * 32, hlf), s1);
    }
    if (col0 + 31 > rowBase) {                    // causal mask (diag blocks)
      #pragma unroll
      for (int r = 0; r < 8; ++r) {
        int row = rowBase + hlf * 8 + r;
        if (col0 + lc      > row) s0[r] = -1e30f;
        if (col0 + 16 + lc > row) s1[r] = -1e30f;
      }
    }
    float scr[8];
    #pragma unroll
    for (int r = 0; r < 8; ++r) {                 // online softmax, fp32
      float mx = fmaxf(s0[r], s1[r]);
      #pragma unroll
      for (int off = 1; off < 16; off <<= 1)
        mx = fmaxf(mx, __shfl_xor(mx, off, 32));  // stay within 16-lane half
      float mnew = fmaxf(mrow[r], mx);
      float sc = exp2f((mrow[r] - mnew) * 1.44269504f);
      float p0 = exp2f((s0[r] - mnew) * 1.44269504f);
      float p1 = exp2f((s1[r] - mnew) * 1.44269504f);
      float ps = p0 + p1;
      #pragma unroll
      for (int off = 1; off < 16; off <<= 1)
        ps += __shfl_xor(ps, off, 32);
      lrow[r] = lrow[r] * sc + ps;
      mrow[r] = mnew;
      scr[r]  = sc;
      s0[r] = p0; s1[r] = p1;
    }
    #pragma unroll
    for (int n = 0; n < 16; ++n)
      #pragma unroll
      for (int r = 0; r < 8; ++r) o[n][r] *= scr[r];
    #pragma unroll
    for (int r = 0; r < 8; ++r) {                 // stage P (C/D -> A layout)
      int rl = hlf * 8 + r;
      Pst[rl * 32 + lc]      = (_Float16)s0[r];
      Pst[rl * 32 + 16 + lc] = (_Float16)s1[r];
    }
    asm volatile("s_wait_dscnt 0" ::: "memory");
    v16h pa = ldfrag(Pst + lc * 32, 0, hlf);
    #pragma unroll
    for (int n = 0; n < 16; ++n) {                // O += P * V  (Vt contiguous)
      const _Float16* Vrow = Vtb + (size_t)(n * 16 + lc) * NLSEQ + col0;
      o[n] = wmma_f16(pa, ldfrag(Vrow, 0, hlf), o[n]);
    }
    asm volatile("s_wait_dscnt 0" ::: "memory");
  }

  #pragma unroll
  for (int r = 0; r < 8; ++r) {                   // normalize by row sum
    float inv = 1.0f / lrow[r];
    #pragma unroll
    for (int n = 0; n < 16; ++n) o[n][r] *= inv;
  }

  // fused fc: facc[l][c] = sum_a Va[l][a] * fc_w[c][a], staged in two halves
  v8f facc[16];
  #pragma unroll
  for (int n = 0; n < 16; ++n)
    #pragma unroll
    for (int r = 0; r < 8; ++r) facc[n][r] = 0.0f;

  _Float16* Ost = &stage[wv][0];                  // 16x128 f16 half-stage
  #pragma unroll
  for (int hh = 0; hh < 2; ++hh) {
    #pragma unroll
    for (int n = 0; n < 8; ++n) {
      int nt = hh * 8 + n;
      #pragma unroll
      for (int r = 0; r < 8; ++r)
        Ost[(hlf * 8 + r) * 128 + n * 16 + lc] = (_Float16)o[nt][r];
    }
    asm volatile("s_wait_dscnt 0" ::: "memory");
    #pragma unroll
    for (int k = 0; k < 4; ++k) {
      v16h ao = ldfrag(Ost + lc * 128, k * 32, hlf);
      #pragma unroll
      for (int n = 0; n < 16; ++n) {
        const _Float16* Wrow = fch + (size_t)(n * 16 + lc) * NAH + hh * 128;
        facc[n] = wmma_f16(ao, ldfrag(Wrow, k * 32, hlf), facc[n]);
      }
    }
    asm volatile("s_wait_dscnt 0" ::: "memory");
  }

  // epilogue: X + (fc + fc_b) * gama
  const float g = gama[0];
  const float* Xb = X + (size_t)bv * NLSEQ * NCH;
  float* Ob = out + (size_t)bv * NLSEQ * NCH;
  #pragma unroll
  for (int n = 0; n < 16; ++n) {
    int c = n * 16 + lc;
    float fb = fcb[c];
    #pragma unroll
    for (int r = 0; r < 8; ++r) {
      int row = rowBase + hlf * 8 + r;
      size_t idx = (size_t)row * NCH + c;
      Ob[idx] = Xb[idx] + (facc[n][r] + fb) * g;
    }
  }
}

extern "C" void kernel_launch(void* const* d_in, const int* in_sizes, int n_in,
                              void* d_out, int out_size, void* d_ws, size_t ws_size,
                              hipStream_t stream) {
  (void)in_sizes; (void)n_in; (void)out_size; (void)ws_size;
  const float* X    = (const float*)d_in[0];
  const float* Qw   = (const float*)d_in[1];
  const float* Kw   = (const float*)d_in[2];
  const float* Vw   = (const float*)d_in[3];
  const float* Qb   = (const float*)d_in[4];
  const float* Kb   = (const float*)d_in[5];
  const float* Vb   = (const float*)d_in[6];
  const float* gama = (const float*)d_in[7];
  const float* fcw  = (const float*)d_in[8];
  const float* fcb  = (const float*)d_in[9];
  float* out = (float*)d_out;

  const size_t szQ = (size_t)NBATCH * NVAR * NLSEQ * NAH;  // f16 elements
  _Float16* Qh  = (_Float16*)d_ws;       // 32 MB
  _Float16* Kh  = Qh + szQ;              // 32 MB
  _Float16* Vt  = Kh + szQ;              // 32 MB (transposed [bv][a][l])
  _Float16* fch = Vt + szQ;              // 128 KB

  cvt_fcw_kernel<<<(NCH * NAH) / 256, 256, 0, stream>>>(fcw, fch);
  dim3 gp(NLSEQ / BMP, NBATCH * NVAR);   // 16 x 64 = 1024 blocks
  proj_kernel<<<gp, 256, 0, stream>>>(X, Qw, Kw, Vw, Qb, Kb, Vb, Qh, Kh, Vt);
  dim3 ga(NLSEQ / BMA, NBATCH * NVAR);   // 8 x 64 = 512 blocks
  attn_kernel<<<ga, 256, 0, stream>>>(X, Qh, Kh, Vt, fch, fcb, gama, out);
}
